// GraphCenteredNetV2_73375221284882
// MI455X (gfx1250) — compile-verified, matrix operand order checked by reading, not run on hardware
//
#include <hip/hip_runtime.h>

#define NN 50000
#define NE 800000
#define NG 64

typedef __attribute__((ext_vector_type(16))) __bf16 v16bf;
typedef __attribute__((ext_vector_type(4)))  __bf16 v4bf;
typedef __attribute__((ext_vector_type(8)))  float  v8f;

// ---------------- utility kernels ----------------

__global__ void zero_u128(uint4* p, int n4) {
  int i = blockIdx.x * blockDim.x + threadIdx.x;
  if (i < n4) p[i] = make_uint4(0u, 0u, 0u, 0u);
}

// WT[n][k] = W[k][n], f32 -> bf16  (W is [K][64] row-major)
__global__ void transpose_w_bf16(const float* __restrict__ W,
                                 __bf16* __restrict__ out, int K) {
  int i = blockIdx.x * blockDim.x + threadIdx.x;
  if (i < K * 64) {
    int n = i / K, k = i % K;
    out[n * K + k] = (__bf16)W[k * 64 + n];
  }
}

// ---------------- fused EdgeConv (gather -> MLP(WMMA) -> relu-max scatter) ----
// One wave handles 16 edges. agg must be zero-initialized; relu(segment_max)
// with empty->0 == u32 atomicMax of max(msg,0) bit patterns (values >= 0).

__launch_bounds__(256)
__global__ void edgeconv_wmma(const float* __restrict__ xin,     // [NN][64]
                              const int*  __restrict__ src,
                              const int*  __restrict__ dst,
                              const __bf16* __restrict__ w1t,    // [64][128] bf16
                              const float*  __restrict__ b1,     // [64]
                              const __bf16* __restrict__ w2t,    // [64][64] bf16
                              const float*  __restrict__ b2,     // [64]
                              unsigned int* __restrict__ agg)    // [NN][64] as u32
{
  __shared__ __attribute__((aligned(16))) __bf16 msg_s[8 * 16 * 128];
  __shared__ __attribute__((aligned(16))) __bf16 hid_s[8 * 16 * 64];
  __shared__ int dst_s[8 * 16];
  __shared__ int src_s[8 * 16];

  const int wave = threadIdx.x >> 5;
  const int lane = threadIdx.x & 31;
  __bf16* msg = msg_s + wave * (16 * 128);
  __bf16* hid = hid_s + wave * (16 * 64);
  int*   dloc = dst_s + wave * 16;
  int*   sloc = src_s + wave * 16;
  const long e0 = ((long)blockIdx.x * 8 + wave) * 16;

  // stage edge indices for this tile into LDS (one global load per lane)
  if (lane < 16) dloc[lane] = dst[e0 + lane];
  else           sloc[lane - 16] = src[e0 + lane - 16];
  asm volatile("s_wait_dscnt 0" ::: "memory");

  // Build message tile with 128-bit gathers:
  //   msg[e][0:64]=x_i (bf16), msg[e][64:128]=x_j-x_i
  for (int i = lane; i < 16 * 16; i += 32) {
    const int e = i >> 4;
    const int f = (i & 15) << 2;
    const int de = dloc[e];
    const int se = sloc[e];
    const float4 xi = *(const float4*)(xin + (long)de * 64 + f);
    const float4 xj = *(const float4*)(xin + (long)se * 64 + f);
    v4bf a, d;
    a[0] = (__bf16)xi.x;  a[1] = (__bf16)xi.y;
    a[2] = (__bf16)xi.z;  a[3] = (__bf16)xi.w;
    d[0] = (__bf16)(xj.x - xi.x);  d[1] = (__bf16)(xj.y - xi.y);
    d[2] = (__bf16)(xj.z - xi.z);  d[3] = (__bf16)(xj.w - xi.w);
    *(v4bf*)(msg + e * 128 + f)      = a;
    *(v4bf*)(msg + e * 128 + 64 + f) = d;
  }
  asm volatile("s_wait_dscnt 0" ::: "memory");  // wave-local LDS visibility

  // ISA 16-bit A layout: lane half selects +8 on K; elems 0..7 -> K=c..c+7,
  // elems 8..15 -> K=c+16..c+23  => two contiguous 16B runs per fragment.
  const int mrow  = lane & 15;
  const int half  = lane >> 4;
  const int koff  = half << 3;   // 0 or 8
  const int ncol  = lane & 15;
  const int mbase = half << 3;   // C-matrix: VGPR r -> row r (lanes 0-15) / r+8

  // ---- GEMM1: [16x128] x [128x64] ----
  v8f acc[4] = {};
  for (int kc = 0; kc < 4; ++kc) {
    const int kbase = kc * 32 + koff;
    v16bf a;
    ((uint4*)&a)[0] = *(const uint4*)(msg + mrow * 128 + kbase);
    ((uint4*)&a)[1] = *(const uint4*)(msg + mrow * 128 + kbase + 16);
    for (int nt = 0; nt < 4; ++nt) {
      const __bf16* wp = w1t + (nt * 16 + ncol) * 128 + kbase;
      v16bf b;
      ((uint4*)&b)[0] = *(const uint4*)wp;
      ((uint4*)&b)[1] = *(const uint4*)(wp + 16);
      acc[nt] = __builtin_amdgcn_wmma_f32_16x16x32_bf16(
          false, a, false, b, (short)0, acc[nt], false, false);
    }
  }

  // bias + ReLU -> hidden tile (bf16) in LDS
  for (int nt = 0; nt < 4; ++nt) {
    const float bias = b1[nt * 16 + ncol];
    for (int r = 0; r < 8; ++r) {
      float v = acc[nt][r] + bias;
      v = v > 0.f ? v : 0.f;
      hid[(mbase + r) * 64 + nt * 16 + ncol] = (__bf16)v;
    }
  }
  asm volatile("s_wait_dscnt 0" ::: "memory");

  // ---- GEMM2: [16x64] x [64x64] ----
  v8f acc2[4] = {};
  for (int kc = 0; kc < 2; ++kc) {
    const int kbase = kc * 32 + koff;
    v16bf a;
    ((uint4*)&a)[0] = *(const uint4*)(hid + mrow * 64 + kbase);
    ((uint4*)&a)[1] = *(const uint4*)(hid + mrow * 64 + kbase + 16);
    for (int nt = 0; nt < 4; ++nt) {
      const __bf16* wp = w2t + (nt * 16 + ncol) * 64 + kbase;
      v16bf b;
      ((uint4*)&b)[0] = *(const uint4*)wp;
      ((uint4*)&b)[1] = *(const uint4*)(wp + 16);
      acc2[nt] = __builtin_amdgcn_wmma_f32_16x16x32_bf16(
          false, a, false, b, (short)0, acc2[nt], false, false);
    }
  }

  // bias + clamp(>=0) + segment-max scatter (u32 max == f32 max for v>=0)
  for (int nt = 0; nt < 4; ++nt) {
    const float bias = b2[nt * 16 + ncol];
    for (int r = 0; r < 8; ++r) {
      float v = acc2[nt][r] + bias;
      v = v > 0.f ? v : 0.f;
      const int node = dloc[mbase + r];
      atomicMax(agg + (long)node * 64 + nt * 16 + ncol, __float_as_uint(v));
    }
  }
}

// ---------------- global max-pool over graphs ----------------
__global__ void pool_max(const float* __restrict__ h,   // [NN][64], all >= 0
                         const int* __restrict__ batch,
                         unsigned int* __restrict__ z)  // [NG][64] zero-init
{
  long i = (long)blockIdx.x * blockDim.x + threadIdx.x;
  if (i < (long)NN * 64) {
    int v = (int)(i >> 6);
    int f = (int)(i & 63);
    int g = batch[v];
    atomicMax(z + g * 64 + f, __float_as_uint(h[i]));
  }
}

// ---------------- decoder MLP 64->32->16->4 (tiny) ----------------
__global__ void decoder_mlp(const float* __restrict__ z,
                            const float* __restrict__ Wd1, const float* __restrict__ bd1,
                            const float* __restrict__ Wd2, const float* __restrict__ bd2,
                            const float* __restrict__ Wd3, const float* __restrict__ bd3,
                            float* __restrict__ out) {
  int g = threadIdx.x;
  if (g >= NG) return;
  float h1[32], h2[16];
  for (int j = 0; j < 32; ++j) {
    float s = bd1[j];
    for (int k = 0; k < 64; ++k) s += z[g * 64 + k] * Wd1[k * 32 + j];
    h1[j] = s > 0.f ? s : 0.f;
  }
  for (int j = 0; j < 16; ++j) {
    float s = bd2[j];
    for (int k = 0; k < 32; ++k) s += h1[k] * Wd2[k * 16 + j];
    h2[j] = s > 0.f ? s : 0.f;
  }
  for (int c = 0; c < 4; ++c) {
    float s = bd3[c];
    for (int k = 0; k < 16; ++k) s += h2[k] * Wd3[k * 4 + c];
    out[g * 4 + c] = s;
  }
}

// ---------------- launch ----------------
extern "C" void kernel_launch(void* const* d_in, const int* in_sizes, int n_in,
                              void* d_out, int out_size, void* d_ws, size_t ws_size,
                              hipStream_t stream) {
  (void)in_sizes; (void)n_in; (void)out_size; (void)ws_size;
  const float* x         = (const float*)d_in[0];
  const int*   ei        = (const int*)d_in[1];     // [2][NE]
  const float* edge_attr = (const float*)d_in[2];
  const int*   batch     = (const int*)d_in[3];
  const float* We1 = (const float*)d_in[4];  const float* be1 = (const float*)d_in[5];
  const float* We2 = (const float*)d_in[6];  const float* be2 = (const float*)d_in[7];
  const float* Wc1 = (const float*)d_in[8];  const float* bc1 = (const float*)d_in[9];
  const float* Wc2 = (const float*)d_in[10]; const float* bc2 = (const float*)d_in[11];
  const float* Wd1 = (const float*)d_in[12]; const float* bd1 = (const float*)d_in[13];
  const float* Wd2 = (const float*)d_in[14]; const float* bd2 = (const float*)d_in[15];
  const float* Wd3 = (const float*)d_in[16]; const float* bd3 = (const float*)d_in[17];

  // workspace layout (16B aligned slices)
  char* ws = (char*)d_ws;
  float*    aggA = (float*)ws;                        // 12.8 MB
  float*    aggB = (float*)(ws + 12800000);           // 12.8 MB
  unsigned* z    = (unsigned*)(ws + 25600000);        // 16 KB
  __bf16*   wt   = (__bf16*)(ws + 25616384);          // 72 KB transposed bf16 weights
  __bf16 *w1t0 = wt,             *w2t0 = wt + 8192;
  __bf16 *w1t1 = wt + 12288,     *w2t1 = wt + 12288 + 8192;
  __bf16 *w1t2 = wt + 24576,     *w2t2 = wt + 24576 + 8192;

  // edge_attr passthrough -> d_out[256:]
  hipMemcpyAsync((float*)d_out + NG * 4, edge_attr,
                 (size_t)NE * 4 * sizeof(float), hipMemcpyDeviceToDevice, stream);

  // pack weights: bf16, transposed [n][k]
  transpose_w_bf16<<<32, 256, 0, stream>>>(We1, w1t0, 128);
  transpose_w_bf16<<<16, 256, 0, stream>>>(We2, w2t0, 64);
  transpose_w_bf16<<<32, 256, 0, stream>>>(Wc1,            w1t1, 128);
  transpose_w_bf16<<<16, 256, 0, stream>>>(Wc2,            w2t1, 64);
  transpose_w_bf16<<<32, 256, 0, stream>>>(Wc1 + 128 * 64, w1t2, 128);
  transpose_w_bf16<<<16, 256, 0, stream>>>(Wc2 + 64 * 64,  w2t2, 64);

  const int* srcP = ei;
  const int* dstP = ei + NE;
  const int EC_BLOCKS = NE / (16 * 8);   // 6250
  const int AGG4 = NN * 64 / 4;          // 800000 uint4

  zero_u128<<<AGG4 / 256, 256, 0, stream>>>((uint4*)aggA, AGG4);
  edgeconv_wmma<<<EC_BLOCKS, 256, 0, stream>>>(x,    srcP, dstP, w1t0, be1,      w2t0, be2,      (unsigned*)aggA);
  zero_u128<<<AGG4 / 256, 256, 0, stream>>>((uint4*)aggB, AGG4);
  edgeconv_wmma<<<EC_BLOCKS, 256, 0, stream>>>(aggA, srcP, dstP, w1t1, bc1,      w2t1, bc2,      (unsigned*)aggB);
  zero_u128<<<AGG4 / 256, 256, 0, stream>>>((uint4*)aggA, AGG4);
  edgeconv_wmma<<<EC_BLOCKS, 256, 0, stream>>>(aggB, srcP, dstP, w1t2, bc1 + 64, w2t2, bc2 + 64, (unsigned*)aggA);

  zero_u128<<<4, 256, 0, stream>>>((uint4*)z, NG * 64 / 4);
  pool_max<<<(NN * 64 + 255) / 256, 256, 0, stream>>>(aggA, batch, z);
  decoder_mlp<<<1, 64, 0, stream>>>((const float*)z, Wd1, bd1, Wd2, bd2, Wd3, bd3, (float*)d_out);
}
